// DeformableConv2d_24429773979920
// MI455X (gfx1250) — matrix operand order repaired and checked by reading, not compiled, not run on hardware
//
#include <hip/hip_runtime.h>
#include <hip/hip_bf16.h>

typedef __attribute__((ext_vector_type(16))) _Float16 v16h;
typedef __attribute__((ext_vector_type(8)))  _Float16 v8h;
typedef __attribute__((ext_vector_type(8)))  float    v8f;

#define Bn    8
#define CIN   128
#define COUT  128
#define Hdim  64
#define Wdim  64
#define HW    4096          // 64*64
#define K2    9
#define NTOT  32768         // B*H*W
#define LDSS  144           // LDS row stride in halves (128 + 16 pad, 288B = 9*32B)

// ---------------------------------------------------------------------------
// Kernel 1: offset conv (B,18,64,64) = conv3x3(x, offset_w) + offset_b
// ---------------------------------------------------------------------------
__global__ __launch_bounds__(256) void offset_conv_kernel(
    const float* __restrict__ x, const float* __restrict__ ow,
    const float* __restrict__ ob, float* __restrict__ offs)
{
    int t = blockIdx.x * 256 + threadIdx.x;
    if (t >= Bn * 18 * HW) return;
    int hw   = t & (HW - 1);
    int rest = t >> 12;
    int oc   = rest % 18;
    int b    = rest / 18;
    int h = hw >> 6, w = hw & 63;

    float acc = ob[oc];
    const float* xb = x  + (size_t)b * CIN * HW;
    const float* wb = ow + (size_t)oc * CIN * 9;
    for (int c = 0; c < CIN; ++c) {
        const float* xc = xb + c * HW;
        const float* wc = wb + c * 9;
        #pragma unroll
        for (int ky = 0; ky < 3; ++ky) {
            int iy = h - 1 + ky;
            if ((unsigned)iy < 64u) {
                #pragma unroll
                for (int kx = 0; kx < 3; ++kx) {
                    int ix = w - 1 + kx;
                    if ((unsigned)ix < 64u)
                        acc += xc[iy * 64 + ix] * wc[ky * 3 + kx];
                }
            }
        }
    }
    offs[t] = acc;
}

// ---------------------------------------------------------------------------
// Kernel 2: pack deform_w (Cout,Cin,3,3) f32 -> Wp[tap][o][c] f16
// ---------------------------------------------------------------------------
__global__ __launch_bounds__(256) void pack_w_kernel(
    const float* __restrict__ dw, _Float16* __restrict__ Wp)
{
    int t = blockIdx.x * 256 + threadIdx.x;
    if (t >= K2 * COUT * CIN) return;
    int kpos = t / (COUT * CIN);
    int rem  = t - kpos * (COUT * CIN);
    int o = rem >> 7;
    int c = rem & 127;
    Wp[t] = (_Float16)dw[(o * CIN + c) * K2 + kpos];
}

// ---------------------------------------------------------------------------
// Kernel 3: fused deformable-sample + WMMA implicit GEMM + BN + SiLU
//   grid = 512 blocks (N tiles of 64), block = 256 threads (8 waves)
// ---------------------------------------------------------------------------
__global__ __launch_bounds__(256) void deform_main_kernel(
    const float* __restrict__ x,   const _Float16* __restrict__ Wp,
    const float* __restrict__ offs, const float* __restrict__ dbias,
    const float* __restrict__ gamma, const float* __restrict__ beta,
    const float* __restrict__ mean,  const float* __restrict__ var,
    float* __restrict__ out)
{
    __shared__ alignas(32) _Float16 s[64 * LDSS];   // samp tile: [j][c], 18 KB

    const int g    = blockIdx.x;
    const int t    = threadIdx.x;
    const int lane = t & 31;
    const int wave = t >> 5;           // Mtile index, 0..7
    const int half = lane >> 4;        // K-half within wmma fragment
    const int r    = lane & 15;

    // sampling thread mapping: j = output column in tile, cgrp = channel group
    const int j    = t & 63;
    const int cgrp = t >> 6;           // 0..3, channels cgrp*32 .. +31
    const int n0   = g * 64;
    const int nj   = n0 + j;
    const int bj   = nj >> 12;
    const int hwj  = nj & (HW - 1);
    const int hj   = hwj >> 6, wj = hwj & 63;

    const v8f vzero = {};
    v8f acc[4] = { vzero, vzero, vzero, vzero };

    for (int kpos = 0; kpos < K2; ++kpos) {
        // ---- bilinear deformable sampling into LDS (f16) ----
        const float dy = offs[((size_t)(bj * 18 + kpos * 2    )) * HW + hwj];
        const float dx = offs[((size_t)(bj * 18 + kpos * 2 + 1)) * HW + hwj];
        const int ky = kpos / 3, kx = kpos - ky * 3;
        const float py = (float)(hj - 1 + ky) + dy;
        const float px = (float)(wj - 1 + kx) + dx;
        const float fy0 = floorf(py), fx0 = floorf(px);
        const int y0 = (int)fy0, x0 = (int)fx0;
        const int y1 = y0 + 1,   x1 = x0 + 1;
        const float wy1 = py - fy0, wx1 = px - fx0;
        const float wy0 = 1.0f - wy1, wx0 = 1.0f - wx1;
        const float vy0 = ((unsigned)y0 < 64u) ? 1.0f : 0.0f;
        const float vy1 = ((unsigned)y1 < 64u) ? 1.0f : 0.0f;
        const float vx0 = ((unsigned)x0 < 64u) ? 1.0f : 0.0f;
        const float vx1 = ((unsigned)x1 < 64u) ? 1.0f : 0.0f;
        const float w00 = wy0 * wx0 * vy0 * vx0;
        const float w01 = wy0 * wx1 * vy0 * vx1;
        const float w10 = wy1 * wx0 * vy1 * vx0;
        const float w11 = wy1 * wx1 * vy1 * vx1;
        const int yc0 = min(max(y0, 0), 63), yc1 = min(max(y1, 0), 63);
        const int xc0 = min(max(x0, 0), 63), xc1 = min(max(x1, 0), 63);
        const int i00 = yc0 * 64 + xc0, i01 = yc0 * 64 + xc1;
        const int i10 = yc1 * 64 + xc0, i11 = yc1 * 64 + xc1;

        const float* xb = x + ((size_t)bj * CIN + cgrp * 32) * HW;
        _Float16* srow = s + j * LDSS + cgrp * 32;
        #pragma unroll 8
        for (int c = 0; c < 32; ++c) {
            const float* xc = xb + c * HW;
            float v = w00 * xc[i00] + w01 * xc[i01]
                    + w10 * xc[i10] + w11 * xc[i11];
            srow[c] = (_Float16)v;
        }
        __syncthreads();

        // ---- WMMA: Mtile = wave, 4 K-subtiles x 4 N-subtiles ----
        const _Float16* wbase = Wp + (size_t)kpos * (COUT * CIN)
                                   + (wave * 16 + r) * CIN;
        #pragma unroll
        for (int kt = 0; kt < 4; ++kt) {
            // A fragment: lane holds row M=r; K = half*16 + 0..15 contiguous
            const _Float16* ap = wbase + kt * 32 + half * 16;
            v16h a = *(const v16h*)ap;

            #pragma unroll
            for (int nt = 0; nt < 4; ++nt) {
                // B fragment: lane holds col N=r, K = half*16 + 0..15 contiguous
                const _Float16* bp = s + (nt * 16 + r) * LDSS
                                       + kt * 32 + half * 16;
                v16h bfrag = *(const v16h*)bp;
                acc[nt] = __builtin_amdgcn_wmma_f32_16x16x32_f16(
                    false, a, false, bfrag, (short)0, acc[nt], false, false);
            }
        }
        __syncthreads();
    }

    // ---- epilogue: bias + BN(inference) + SiLU ----
    #pragma unroll
    for (int nt = 0; nt < 4; ++nt) {
        const int col = nt * 16 + r;
        const int ng  = n0 + col;
        const int b   = ng >> 12;
        const int hw  = ng & (HW - 1);
        #pragma unroll
        for (int i = 0; i < 8; ++i) {
            const int o = wave * 16 + half * 8 + i;   // C/D layout: lane>=16 -> M=8+i
            float v  = acc[nt][i] + dbias[o];
            float sc = gamma[o] * rsqrtf(var[o] + 1e-5f);
            v = (v - mean[o]) * sc + beta[o];
            float sig = 1.0f / (1.0f + __expf(-v));
            out[((size_t)(b * COUT + o)) * HW + hw] = v * sig;
        }
    }
}

// ---------------------------------------------------------------------------
extern "C" void kernel_launch(void* const* d_in, const int* in_sizes, int n_in,
                              void* d_out, int out_size, void* d_ws, size_t ws_size,
                              hipStream_t stream) {
    const float* x        = (const float*)d_in[0];
    const float* offset_w = (const float*)d_in[1];
    const float* offset_b = (const float*)d_in[2];
    const float* deform_w = (const float*)d_in[3];
    const float* deform_b = (const float*)d_in[4];
    const float* gamma    = (const float*)d_in[5];
    const float* beta     = (const float*)d_in[6];
    const float* rmean    = (const float*)d_in[7];
    const float* rvar     = (const float*)d_in[8];
    float* out = (float*)d_out;

    float*     offs = (float*)d_ws;                               // 2,359,296 B
    _Float16*  Wp   = (_Float16*)((char*)d_ws +
                         (size_t)Bn * 18 * HW * sizeof(float));   // +294,912 B

    {
        int n = Bn * 18 * HW;
        offset_conv_kernel<<<(n + 255) / 256, 256, 0, stream>>>(x, offset_w, offset_b, offs);
    }
    {
        int n = K2 * COUT * CIN;
        pack_w_kernel<<<(n + 255) / 256, 256, 0, stream>>>(deform_w, Wp);
    }
    deform_main_kernel<<<NTOT / 64, 256, 0, stream>>>(
        x, Wp, offs, deform_b, gamma, beta, rmean, rvar, out);
}
